// GaussianRasterizerIndexed_35038343201333
// MI455X (gfx1250) — matrix operand order repaired and checked by reading, not compiled, not run on hardware
//
#include <hip/hip_runtime.h>

#define AS1 __attribute__((address_space(1)))
#define AS3 __attribute__((address_space(3)))

typedef _Float16 v16h __attribute__((ext_vector_type(16)));
typedef _Float16 v8h  __attribute__((ext_vector_type(8)));
typedef float    v8f  __attribute__((ext_vector_type(8)));

#define NG      4096
#define NCHUNK  128          // 4096 gaussians / 32 per chunk
#define IMG_H   80
#define IMG_W   80
#define NPIX    (IMG_H*IMG_W)

// FOVX=FOVY=pi/2 -> tan = 1, focal = 80/2 = 40
#define TANX 1.0f
#define TANY 1.0f
#define FX   40.0f
#define FY   40.0f

// ---------------------------------------------------------------------------
// CDNA5 async global->LDS staging (ASYNCcnt tracked), hedged builtin/asm
// ---------------------------------------------------------------------------
__device__ __forceinline__ void async_load_dword(const unsigned* gbase,
                                                 unsigned dword_off,
                                                 unsigned* ldsptr) {
#if __has_builtin(__builtin_amdgcn_global_load_async_to_lds_b32)
  __builtin_amdgcn_global_load_async_to_lds_b32(
      (AS1 int*)(gbase + dword_off), (AS3 int*)ldsptr, 0, 0);
#else
  unsigned loff = (unsigned)(unsigned long long)(AS3 unsigned*)ldsptr;
  unsigned voff = dword_off * 4u;
  asm volatile("global_load_async_to_lds_b32 %0, %1, %2"
               :: "v"(loff), "v"(voff), "s"((unsigned long long)gbase)
               : "memory");
#endif
}

template <int N>
__device__ __forceinline__ void wait_asynccnt() {
#if __has_builtin(__builtin_amdgcn_s_wait_asynccnt)
  __builtin_amdgcn_s_wait_asynccnt(N);
#else
  asm volatile("s_wait_asynccnt %0" :: "i"(N) : "memory");
#endif
}

// ---------------------------------------------------------------------------
// Kernel 1: per-gaussian preprocessing.
// Emits: radii (into d_out tail), sort key (depth|idx), packed 8-dword record:
//   [gx, gy, conA, conB, conC, op_eff, half2(r,g), half2(b,0)]
// ---------------------------------------------------------------------------
__global__ __launch_bounds__(256) void preprocess_kernel(
    const float* __restrict__ means, const float* __restrict__ sh,
    const int* __restrict__ sh_idx, const int* __restrict__ g_idx,
    const float* __restrict__ opac, const float* __restrict__ scales,
    const float* __restrict__ sfac, const float* __restrict__ rots,
    const float* __restrict__ E, unsigned long long* __restrict__ keys,
    unsigned* __restrict__ params, int* __restrict__ radii_out) {
  int i = blockIdx.x * blockDim.x + threadIdx.x;
  if (i >= NG) return;

  // extrinsic (row-vector convention: p_view = m @ R3 + t3)
  float R00=E[0],  R01=E[1],  R02=E[2];
  float R10=E[4],  R11=E[5],  R12=E[6];
  float R20=E[8],  R21=E[9],  R22=E[10];
  float t0=E[12],  t1=E[13],  t2=E[14];

  float m0=means[3*i], m1=means[3*i+1], m2=means[3*i+2];
  float tx = m0*R00 + m1*R10 + m2*R20 + t0;
  float ty = m0*R01 + m1*R11 + m2*R21 + t1;
  float tz = m0*R02 + m1*R12 + m2*R22 + t2;
  bool in_front = tz > 0.2f;
  float tzs = in_front ? tz : 1.0f;
  float gx = ((tx/(tzs*TANX) + 1.f)*(float)IMG_W - 1.f)*0.5f;
  float gy = ((ty/(tzs*TANY) + 1.f)*(float)IMG_H - 1.f)*0.5f;

  int gi = g_idx[i];
  float sf = sfac[i];
  float s0 = scales[3*gi]*sf, s1 = scales[3*gi+1]*sf, s2 = scales[3*gi+2]*sf;
  float qw=rots[4*gi], qx=rots[4*gi+1], qy=rots[4*gi+2], qz=rots[4*gi+3];
  float qinv = 1.f/sqrtf(qw*qw+qx*qx+qy*qy+qz*qz);
  qw*=qinv; qx*=qinv; qy*=qinv; qz*=qinv;
  float Rm00 = 1.f-2.f*(qy*qy+qz*qz), Rm01 = 2.f*(qx*qy-qw*qz), Rm02 = 2.f*(qx*qz+qw*qy);
  float Rm10 = 2.f*(qx*qy+qw*qz), Rm11 = 1.f-2.f*(qx*qx+qz*qz), Rm12 = 2.f*(qy*qz-qw*qx);
  float Rm20 = 2.f*(qx*qz-qw*qy), Rm21 = 2.f*(qy*qz+qw*qx), Rm22 = 1.f-2.f*(qx*qx+qy*qy);
  float d0=s0*s0, d1=s1*s1, d2=s2*s2;
  // C = Rm diag(d) Rm^T (symmetric)
  float C00 = Rm00*Rm00*d0 + Rm01*Rm01*d1 + Rm02*Rm02*d2;
  float C01 = Rm00*Rm10*d0 + Rm01*Rm11*d1 + Rm02*Rm12*d2;
  float C02 = Rm00*Rm20*d0 + Rm01*Rm21*d1 + Rm02*Rm22*d2;
  float C11 = Rm10*Rm10*d0 + Rm11*Rm11*d1 + Rm12*Rm12*d2;
  float C12 = Rm10*Rm20*d0 + Rm11*Rm21*d1 + Rm12*Rm22*d2;
  float C22 = Rm20*Rm20*d0 + Rm21*Rm21*d1 + Rm22*Rm22*d2;
  // Wm = R3^T
  float W00=R00, W01=R10, W02=R20;
  float W10=R01, W11=R11, W12=R21;
  float W20=R02, W21=R12, W22=R22;
  // Tm = Wm * C
  float T00 = W00*C00 + W01*C01 + W02*C02;
  float T01 = W00*C01 + W01*C11 + W02*C12;
  float T02 = W00*C02 + W01*C12 + W02*C22;
  float T10 = W10*C00 + W11*C01 + W12*C02;
  float T11 = W10*C01 + W11*C11 + W12*C12;
  float T12 = W10*C02 + W11*C12 + W12*C22;
  float T20 = W20*C00 + W21*C01 + W22*C02;
  float T21 = W20*C01 + W21*C11 + W22*C12;
  float T22 = W20*C02 + W21*C12 + W22*C22;
  // V = Tm * Wm^T
  float V00 = T00*W00 + T01*W01 + T02*W02;
  float V01 = T00*W10 + T01*W11 + T02*W12;
  float V02 = T00*W20 + T01*W21 + T02*W22;
  float V11 = T10*W10 + T11*W11 + T12*W12;
  float V12 = T10*W20 + T11*W21 + T12*W22;
  float V22 = T20*W20 + T21*W21 + T22*W22;

  float limx = 1.3f*TANX, limy = 1.3f*TANY;
  float txc = fminf(limx, fmaxf(-limx, tx/tzs))*tzs;
  float tyc = fminf(limy, fmaxf(-limy, ty/tzs))*tzs;
  float J00 = FX/tzs, J02 = -FX*txc/(tzs*tzs);
  float J11 = FY/tzs, J12 = -FY*tyc/(tzs*tzs);
  float c00 = J00*J00*V00 + 2.f*J00*J02*V02 + J02*J02*V22;
  float c01 = J00*J11*V01 + J00*J12*V02 + J02*J11*V12 + J02*J12*V22;
  float c11 = J11*J11*V11 + 2.f*J11*J12*V12 + J12*J12*V22;
  float a = c00 + 0.3f, b = c01, cco = c11 + 0.3f;
  float det = a*cco - b*b;
  bool valid = in_front && (det > 0.f);
  float dets = (det != 0.f) ? det : 1.f;
  float ica = cco/dets, icb = -b/dets, icc = a/dets;
  float mid = 0.5f*(a + cco);
  float lam1 = mid + sqrtf(fmaxf(0.1f, mid*mid - det));
  radii_out[i] = valid ? (int)ceilf(3.f*sqrtf(lam1)) : 0;

  // campos = inv(extrinsic)[3,:3] = -t3 @ R3^{-1}
  float det3 = R00*(R11*R22-R12*R21) - R01*(R10*R22-R12*R20) + R02*(R10*R21-R11*R20);
  float idt = 1.f/det3;
  float i00 = (R11*R22 - R12*R21)*idt, i01 = (R02*R21 - R01*R22)*idt, i02 = (R01*R12 - R02*R11)*idt;
  float i10 = (R12*R20 - R10*R22)*idt, i11 = (R00*R22 - R02*R20)*idt, i12 = (R02*R10 - R00*R12)*idt;
  float i20 = (R10*R21 - R11*R20)*idt, i21 = (R01*R20 - R00*R21)*idt, i22 = (R00*R11 - R01*R10)*idt;
  float cpx = -(t0*i00 + t1*i10 + t2*i20);
  float cpy = -(t0*i01 + t1*i11 + t2*i21);
  float cpz = -(t0*i02 + t1*i12 + t2*i22);

  float dx = m0-cpx, dy = m1-cpy, dz = m2-cpz;
  float dn = 1.f/sqrtf(dx*dx+dy*dy+dz*dz);
  float x = dx*dn, y = dy*dn, z = dz*dn;
  float xx=x*x, yy=y*y, zz=z*z, xy=x*y, yz=y*z, xz=x*z;
  float bfn[16] = {
    0.28209479177387814f,
    -0.4886025119029199f*y,
     0.4886025119029199f*z,
    -0.4886025119029199f*x,
     1.0925484305920792f*xy,
    -1.0925484305920792f*yz,
     0.31539156525252005f*(2.f*zz-xx-yy),
    -1.0925484305920792f*xz,
     0.5462742152960396f*(xx-yy),
    -0.5900435899266435f*y*(3.f*xx-yy),
     2.890611442640554f*xy*z,
    -0.4570457994644658f*y*(4.f*zz-xx-yy),
     0.3731763325901154f*z*(2.f*zz-3.f*xx-3.f*yy),
    -0.4570457994644658f*x*(4.f*zz-xx-yy),
     1.445305721320277f*z*(xx-yy),
    -0.5900435899266435f*x*(xx-yy-3.f*zz)
  };
  const float* shp = sh + sh_idx[i]*48;   // (16,3) per gaussian
  float col[3];
  #pragma unroll
  for (int c = 0; c < 3; ++c) {
    float s = 0.f;
    #pragma unroll
    for (int j = 0; j < 16; ++j) s += bfn[j]*shp[j*3+c];
    col[c] = fmaxf(s + 0.5f, 0.f);
  }

  float op = valid ? opac[i] : 0.f;   // invalid -> alpha 0 -> no contribution
  float depth = valid ? tz : __builtin_huge_valf();
  keys[i] = ((unsigned long long)__float_as_uint(depth) << 32) | (unsigned)i;

  unsigned* P = params + i*8;
  P[0] = __float_as_uint(gx);
  P[1] = __float_as_uint(gy);
  P[2] = __float_as_uint(ica);
  P[3] = __float_as_uint(icb);
  P[4] = __float_as_uint(icc);
  P[5] = __float_as_uint(op);
  unsigned hr = (unsigned)__builtin_bit_cast(unsigned short, (_Float16)col[0]);
  unsigned hg = (unsigned)__builtin_bit_cast(unsigned short, (_Float16)col[1]);
  unsigned hb = (unsigned)__builtin_bit_cast(unsigned short, (_Float16)col[2]);
  P[6] = hr | (hg << 16);
  P[7] = hb;
}

// ---------------------------------------------------------------------------
// Kernel 2: single-workgroup bitonic sort of 4096 (depth|idx) keys in LDS,
// then gather the 8-dword records into sorted order.
// ---------------------------------------------------------------------------
__global__ __launch_bounds__(1024) void sort_gather_kernel(
    const unsigned long long* __restrict__ keys,
    const unsigned* __restrict__ params,
    unsigned* __restrict__ sorted) {
  __shared__ unsigned long long sk[NG];
  for (int i = threadIdx.x; i < NG; i += 1024) sk[i] = keys[i];
  __syncthreads();
  for (int k = 2; k <= NG; k <<= 1) {
    for (int j = k >> 1; j > 0; j >>= 1) {
      for (int i = threadIdx.x; i < NG; i += 1024) {
        int ixj = i ^ j;
        if (ixj > i) {
          unsigned long long va = sk[i], vb = sk[ixj];
          bool up = (i & k) == 0;
          if (up ? (va > vb) : (va < vb)) { sk[i] = vb; sk[ixj] = va; }
        }
      }
      __syncthreads();
    }
  }
  for (int r = threadIdx.x; r < NG; r += 1024) {
    int idx = (int)(sk[r] & 0xFFFFFFFFu);
    #pragma unroll
    for (int d = 0; d < 8; ++d) sorted[r*8 + d] = params[idx*8 + d];
  }
}

// ---------------------------------------------------------------------------
// Kernel 3: tiled rasterizer. 16x16 pixel tile per workgroup (256 thr, 8 waves).
// Double-buffered async LDS staging of 32-gaussian chunks; exact per-pixel
// transmittance scan with uniform early exit; blend via v_wmma_f32_16x16x32_f16.
// ---------------------------------------------------------------------------
__global__ __launch_bounds__(256) void raster_kernel(
    const unsigned* __restrict__ sorted, const float* __restrict__ bg,
    float* __restrict__ outc) {
  __shared__ unsigned stage[2][256];      // 32 gaussians * 8 dwords
  __shared__ _Float16 wtile[256][32];     // per-pixel weights for one chunk
  __shared__ float    tT[256];            // final transmittance per pixel

  const int t    = threadIdx.x;
  const int lane = t & 31;
  const int wid  = t >> 5;
  const int txp  = t & 15;                // pixel x within tile
  const int typ  = t >> 4;                // pixel y within tile
  const float px = (float)(blockIdx.x*16 + txp);
  const float py = (float)(blockIdx.y*16 + typ);

  float T = 1.0f;
  bool done = false;
  v8f acc0 = {}, acc1 = {};               // two 16x16 D tiles per wave

  // stage chunk 0
  async_load_dword(sorted, (unsigned)t, &stage[0][t]);

  for (int c = 0; c < NCHUNK; ++c) {
    if (c + 1 < NCHUNK) {
      async_load_dword(sorted, (unsigned)((c+1)*256 + t), &stage[(c+1)&1][t]);
      wait_asynccnt<1>();                 // chunk c complete, c+1 in flight
    } else {
      wait_asynccnt<0>();
    }
    __syncthreads();                      // all waves' staging visible

    const unsigned* buf = &stage[c & 1][0];

    // ---- per-pixel sequential scan over 32 sorted gaussians (exact) ----
    for (int j = 0; j < 32; ++j) {
      float w = 0.f;
      if (!done) {
        float ggx = __uint_as_float(buf[j*8+0]);
        float ggy = __uint_as_float(buf[j*8+1]);
        float ca  = __uint_as_float(buf[j*8+2]);
        float cb  = __uint_as_float(buf[j*8+3]);
        float cc  = __uint_as_float(buf[j*8+4]);
        float op  = __uint_as_float(buf[j*8+5]);
        float dx = ggx - px, dy = ggy - py;
        float power = -0.5f*(ca*dx*dx + cc*dy*dy) - cb*dx*dy;
        float a = fminf(0.99f, op * expf(fminf(power, 0.f)));
        bool keep = (power <= 0.f) && (a >= (1.f/255.f));
        a = keep ? a : 0.f;
        float Tt = T * (1.f - a);
        if (Tt >= 1e-4f) { w = a * T; T = Tt; }   // inc gate, exact
        else             { done = true; }          // T stays = T_final
      }
      wtile[t][j] = (_Float16)w;
    }
    int nact = __syncthreads_count(done ? 0 : 1);  // barrier + uniform count

    // ---- WMMA blend: D(16px x 16ch) += W(16x32) * Col(32x16, 3 live cols) ----
    {
      const int M  = lane & 15;
      const int hi = lane >> 4;
      const _Float16* r0 = &wtile[(2*wid)*16 + M][0];
      const _Float16* r1 = r0 + 16*32;
      // 16-bit A layout: lanes<16 hold K 0..7 / 16..23; lanes>=16 hold +8
      v8h a0l = *(const v8h*)(r0 + 8*hi);
      v8h a0h = *(const v8h*)(r0 + 16 + 8*hi);
      v8h a1l = *(const v8h*)(r1 + 8*hi);
      v8h a1h = *(const v8h*)(r1 + 16 + 8*hi);
      v16h A0{}, A1{};
      #pragma unroll
      for (int e = 0; e < 8; ++e) {
        A0[e] = a0l[e]; A0[e+8] = a0h[e];
        A1[e] = a1l[e]; A1[e+8] = a1h[e];
      }
      // B: row-striped across lanes; column = lane&15 (channels 0..2 live)
      const int nc = lane & 15;
      v16h B{};
      #pragma unroll
      for (int e = 0; e < 16; ++e) {
        int K = (e & 7) + 8*hi + 16*(e >> 3);
        unsigned short hbits = 0;
        if (nc < 3) {
          unsigned cw = buf[K*8 + 6 + (nc >> 1)];
          hbits = (unsigned short)((nc & 1) ? (cw >> 16) : (cw & 0xFFFFu));
        }
        B[e] = __builtin_bit_cast(_Float16, hbits);
      }
      acc0 = __builtin_amdgcn_wmma_f32_16x16x32_f16(false, A0, false, B,
                                                    (short)0, acc0, false, false);
      acc1 = __builtin_amdgcn_wmma_f32_16x16x32_f16(false, A1, false, B,
                                                    (short)0, acc1, false, false);
    }
    __syncthreads();                      // wtile/stage reusable
    if (nact == 0) break;                 // uniform early exit (exact)
  }

  tT[t] = T;                              // T_final per pixel
  __syncthreads();

  // writeout: D layout: lane holds column N=lane&15, rows M=r+8*(lane>>4)
  const int nch = lane & 15;
  const int hi  = lane >> 4;
  if (nch < 3) {
    float bgc = bg[nch];
    #pragma unroll
    for (int r = 0; r < 8; ++r) {
      int Mr = r + 8*hi;
      {
        int mt = 2*wid;
        int q  = mt*16 + Mr;
        int xg = blockIdx.x*16 + Mr, yg = blockIdx.y*16 + mt;
        outc[nch*NPIX + yg*IMG_W + xg] = acc0[r] + tT[q]*bgc;
      }
      {
        int mt = 2*wid + 1;
        int q  = mt*16 + Mr;
        int xg = blockIdx.x*16 + Mr, yg = blockIdx.y*16 + mt;
        outc[nch*NPIX + yg*IMG_W + xg] = acc1[r] + tT[q]*bgc;
      }
    }
  }
}

// ---------------------------------------------------------------------------
extern "C" void kernel_launch(void* const* d_in, const int* in_sizes, int n_in,
                              void* d_out, int out_size, void* d_ws, size_t ws_size,
                              hipStream_t stream) {
  (void)in_sizes; (void)n_in; (void)out_size; (void)ws_size;
  const float* means  = (const float*)d_in[0];
  const float* sh     = (const float*)d_in[1];
  const int*   sh_idx = (const int*)d_in[2];
  const int*   g_idx  = (const int*)d_in[3];
  const float* opac   = (const float*)d_in[4];
  const float* scales = (const float*)d_in[5];
  const float* sfac   = (const float*)d_in[6];
  const float* rots   = (const float*)d_in[7];
  const float* extr   = (const float*)d_in[8];
  const float* bg     = (const float*)d_in[9];

  char* ws = (char*)d_ws;
  unsigned long long* keys   = (unsigned long long*)ws;            // 32 KB
  unsigned*           params = (unsigned*)(ws + NG*8);             // 128 KB
  unsigned*           sorted = (unsigned*)(ws + NG*8 + NG*32);     // 128 KB

  float* outc  = (float*)d_out;                 // (3,80,80)
  int*   radii = (int*)d_out + 3*NPIX;          // 4096 int32 (bit-exact)

  preprocess_kernel<<<NG/256, 256, 0, stream>>>(means, sh, sh_idx, g_idx, opac,
                                                scales, sfac, rots, extr,
                                                keys, params, radii);
  sort_gather_kernel<<<1, 1024, 0, stream>>>(keys, params, sorted);
  raster_kernel<<<dim3(IMG_W/16, IMG_H/16), 256, 0, stream>>>(sorted, bg, outc);
}
